// _AttenAsppBlock_27075473834455
// MI455X (gfx1250) — matrix operand order repaired and checked by reading, compiled, not run on hardware
//
#include <hip/hip_runtime.h>

// ---------------------------------------------------------------------------
// CDNA5 (gfx1250) implementation: bf16 WMMA GEMMs + f32 VALU attention/softmax
// ---------------------------------------------------------------------------

typedef __attribute__((ext_vector_type(16))) __bf16 v16bf;
typedef __attribute__((ext_vector_type(8)))  __bf16 v8bf;
typedef __attribute__((ext_vector_type(4)))  __bf16 v4bf;
typedef __attribute__((ext_vector_type(2)))  __bf16 v2bf;
typedef __attribute__((ext_vector_type(8)))  float  v8f;

#define HW   4096   // 64*64
#define WDIM 64

// ---------------------------------------------------------------------------
// Tiled GEMM:  Out[b][m][n] = sum_k W[m][k] * X[b][k][n] + bias[m]
//
// Block tile 128(M) x 64(N), K-step 32, 256 threads = 8 waves (4M x 2N).
// Each wave owns a 32x32 output tile = 4 f32 WMMA accumulators
// (v_wmma_f32_16x16x32_bf16), i.e. 4 WMMAs per K-step per wave.
//
// M/K/N are template constants so all address math strength-reduces; staging
// uses per-thread pointers bumped by the K-step stride (no per-iteration
// 64-bit multiplies). Global loads for step k+1 are issued between the
// LDS-store barrier and the compute barrier of step k (latency overlap).
//
// BTRANS=true  : X is [b][N][K] bf16 (pre-transposed) -> contiguous v8bf
//                staging, zero transpose cost (used by w2 reading h1T).
// OTRANS=true  : Out is [b][N][M] bf16 -> epilogue packs the 8 contiguous-M
//                accumulator lanes into one v8bf 16B store (used by w1).
// ---------------------------------------------------------------------------
template <int M, int K, int N, typename TIN, bool BTRANS, bool LEAKY,
          typename TOUT, bool OTRANS>
__global__ __launch_bounds__(256) void gemm_wmma(
    const float* __restrict__ W,     // [M][K] row-major
    const TIN*   __restrict__ X,     // [b][K][N]  (or [b][N][K] if BTRANS)
    const float* __restrict__ bias,  // [M]
    TOUT*        __restrict__ Out)   // [b][M][N]  (or [b][N][M] if OTRANS)
{
  const int b  = blockIdx.z;
  const int m0 = blockIdx.y * 128;
  const int n0 = blockIdx.x * 64;
  const TIN* Xb = X + (size_t)b * K * N;

  // +8 padding: rows 80B apart (16B aligned), spreads LDS banks.
  __shared__ __bf16 As[128][40];  // [m][k]
  __shared__ __bf16 Bs[64][40];   // [n][k]

  const int tid    = threadIdx.x;
  const int lane   = tid & 31;
  const int waveId = tid >> 5;
  const int wm     = (waveId & 3) * 32;   // wave M sub-tile
  const int wn     = (waveId >> 2) * 32;  // wave N sub-tile

  // CDNA5 WMMA wave32 fragment addressing:
  //  A 16x32 bf16: lane L -> row M=L&15, K halves kA..kA+7 / 16+kA..16+kA+7
  //  B 32x16 bf16: lane L -> col N=L&15, K halves kB..kB+7 / 16+kB..16+kB+7
  const int fm = lane & 15;
  const int kA = (lane >> 4) * 8;
  const int kB = (lane >> 4) * 8;

  v8f z = {};
  v8f acc[2][2] = {{z, z}, {z, z}};

  // ---- per-thread staging pointers (bumped by K-step stride) ----
  const float* aPtr[4];
#pragma unroll
  for (int i = 0; i < 4; ++i) {
    int u = tid + i * 256;                  // 1024 float4 units: 128 rows x 8
    int r = u >> 3, c = (u & 7) * 4;
    aPtr[i] = W + (size_t)(m0 + r) * K + c;
  }
  const TIN* bPtr[4];
  if constexpr (BTRANS) {
    bPtr[0] = Xb + (size_t)(n0 + (tid & 63)) * K + (tid >> 6) * 8;
  } else {
#pragma unroll
    for (int j = 0; j < 4; ++j) {           // 1024 k-pair units: 16 x 64
      int u = tid + j * 256;
      int n = u & 63, k = (u >> 6) * 2;
      bPtr[j] = Xb + (size_t)k * N + n0 + n;
    }
  }

  // ---- staging registers (pipelined) ----
  float4 aw[4];
  float  bw[8];
  v8bf   bt;

  auto loadA = [&]() {
#pragma unroll
    for (int i = 0; i < 4; ++i) {
      aw[i] = *(const float4*)aPtr[i];
      aPtr[i] += 32;
    }
  };
  auto loadB = [&]() {
    if constexpr (BTRANS) {
      bt = *(const v8bf*)bPtr[0];
      bPtr[0] += 32;
    } else {
#pragma unroll
      for (int j = 0; j < 4; ++j) {
        bw[2 * j]     = (float)bPtr[j][0];
        bw[2 * j + 1] = (float)bPtr[j][N];
        bPtr[j] += (size_t)32 * N;
      }
    }
  };
  auto storeAB = [&]() {
#pragma unroll
    for (int i = 0; i < 4; ++i) {
      int u = tid + i * 256;
      int r = u >> 3, c = (u & 7) * 4;
      v4bf p;
      p[0] = (__bf16)aw[i].x; p[1] = (__bf16)aw[i].y;
      p[2] = (__bf16)aw[i].z; p[3] = (__bf16)aw[i].w;
      *(v4bf*)&As[r][c] = p;                // ds_store_b64
    }
    if constexpr (BTRANS) {
      int n = tid & 63, kg = (tid >> 6) * 8;
      *(v8bf*)&Bs[n][kg] = bt;              // ds_store_b128
    } else {
#pragma unroll
      for (int j = 0; j < 4; ++j) {
        int u = tid + j * 256;
        int n = u & 63, k = (u >> 6) * 2;
        v2bf p;
        p[0] = (__bf16)bw[2 * j]; p[1] = (__bf16)bw[2 * j + 1];
        *(v2bf*)&Bs[n][k] = p;              // ds_store_b32
      }
    }
  };

  union Frag { v16bf v; v8bf h[2]; };

  loadA();
  loadB();
#pragma unroll 1
  for (int k0 = 0; k0 < K; k0 += 32) {
    __syncthreads();           // previous step's fragment reads done
    storeAB();
    if (k0 + 32 < K) {         // overlap next tile's HBM latency with WMMAs
      loadA();
      loadB();
    }
    __syncthreads();           // LDS tile visible

    Frag fa0, fa1, fb0, fb1;
    fa0.h[0] = *(const v8bf*)&As[wm + fm][kA];
    fa0.h[1] = *(const v8bf*)&As[wm + fm][16 + kA];
    fa1.h[0] = *(const v8bf*)&As[wm + 16 + fm][kA];
    fa1.h[1] = *(const v8bf*)&As[wm + 16 + fm][16 + kA];
    fb0.h[0] = *(const v8bf*)&Bs[wn + fm][kB];
    fb0.h[1] = *(const v8bf*)&Bs[wn + fm][kB + 16];
    fb1.h[0] = *(const v8bf*)&Bs[wn + 16 + fm][kB];
    fb1.h[1] = *(const v8bf*)&Bs[wn + 16 + fm][kB + 16];

    acc[0][0] = __builtin_amdgcn_wmma_f32_16x16x32_bf16(false, fa0.v, false, fb0.v,
                                                        (short)0, acc[0][0], false, false);
    acc[0][1] = __builtin_amdgcn_wmma_f32_16x16x32_bf16(false, fa0.v, false, fb1.v,
                                                        (short)0, acc[0][1], false, false);
    acc[1][0] = __builtin_amdgcn_wmma_f32_16x16x32_bf16(false, fa1.v, false, fb0.v,
                                                        (short)0, acc[1][0], false, false);
    acc[1][1] = __builtin_amdgcn_wmma_f32_16x16x32_bf16(false, fa1.v, false, fb1.v,
                                                        (short)0, acc[1][1], false, false);
  }

  // Epilogue. C/D layout: VGPR r holds M = r + 8*(lane>>4), N = lane&15,
  // so the 8 accumulator entries cover 8 *consecutive* M rows.
  const int mh = (lane >> 4) * 8;
#pragma unroll
  for (int ia = 0; ia < 2; ++ia) {
    const int mbase = m0 + wm + ia * 16 + mh;
    float4 bb0 = *(const float4*)&bias[mbase];
    float4 bb1 = *(const float4*)&bias[mbase + 4];
    float bl[8] = {bb0.x, bb0.y, bb0.z, bb0.w, bb1.x, bb1.y, bb1.z, bb1.w};
#pragma unroll
    for (int ib = 0; ib < 2; ++ib) {
      const int n = n0 + wn + ib * 16 + fm;
      if constexpr (OTRANS) {
        v8bf p;
#pragma unroll
        for (int r = 0; r < 8; ++r) {
          float v = acc[ia][ib][r] + bl[r];
          if (LEAKY) v = (v >= 0.f) ? v : 0.01f * v;
          p[r] = (__bf16)v;
        }
        *(v8bf*)&Out[((size_t)b * N + n) * M + mbase] = p;  // one b128 store
      } else {
#pragma unroll
        for (int r = 0; r < 8; ++r) {
          float v = acc[ia][ib][r] + bl[r];
          if (LEAKY) v = (v >= 0.f) ? v : 0.01f * v;
          Out[(size_t)b * M * N + (size_t)(mbase + r) * N + n] = (TOUT)v;
        }
      }
    }
  }
}

// ---------------------------------------------------------------------------
// 9-neighbor dilated attention (d=3), replicating reference semantics:
//  - K9/V9 zero-padded: OOB neighbor -> logit 0 (participates in softmax),
//    V contribution 0
//  - masked = (nomask != 0) = (mask == 0) for in-bounds neighbors -> NEG logit
//  - nm batch index is bh % 8 (jnp.tile ordering quirk of the reference)
//  - valid = any(~masked); invalid -> P = 0
// QK buffer: [16][256][HW] (Q = ch 0..127, K = ch 128..255); V: [16][128][HW]
// ---------------------------------------------------------------------------
__global__ __launch_bounds__(256) void attention_kernel(
    const float* __restrict__ QK, const float* __restrict__ V,
    const float* __restrict__ mask, float* __restrict__ att)
{
  const int bh  = blockIdx.y;
  const int pos = blockIdx.x * 256 + threadIdx.x;
  const int y = pos >> 6, x = pos & 63;

  const float* Qp = QK + (size_t)bh * 256 * HW;
  const float* Kp = Qp + 128 * HW;
  const float* Vp = V + (size_t)bh * 128 * HW;
  const float* mb = mask + (size_t)(bh & 7) * HW;  // reference tile() ordering

  bool  inb[9];
  int   npos[9];
  bool  msk[9];
  float lg[9];

#pragma unroll
  for (int n = 0; n < 9; ++n) {
    int yy = y + (n / 3 - 1) * 3;
    int xx = x + (n % 3 - 1) * 3;
    inb[n]  = (yy >= 0) && (yy < WDIM) && (xx >= 0) && (xx < WDIM);
    npos[n] = inb[n] ? (yy * WDIM + xx) : pos;  // safe address when OOB
    msk[n]  = inb[n] && (mb[npos[n]] == 0.0f);
    lg[n]   = 0.0f;
  }

  for (int c = 0; c < 128; ++c) {
    float q = Qp[(size_t)c * HW + pos];
    const float* Kc = Kp + (size_t)c * HW;
#pragma unroll
    for (int n = 0; n < 9; ++n)
      lg[n] += q * Kc[npos[n]];
  }

  const float scale = 0.0625f;  // 256^-0.5
  bool valid = false;
  float mx = -3.0e38f;
#pragma unroll
  for (int n = 0; n < 9; ++n) {
    lg[n] = inb[n] ? (msk[n] ? -1.0e30f : lg[n] * scale) : 0.0f;
    valid = valid || !msk[n];  // ~masked: OOB or mask!=0
    mx = fmaxf(mx, lg[n]);
  }

  float s = 0.0f;
  float P[9];
#pragma unroll
  for (int n = 0; n < 9; ++n) { P[n] = __expf(lg[n] - mx); s += P[n]; }
  float inv = valid ? (1.0f / s) : 0.0f;
#pragma unroll
  for (int n = 0; n < 9; ++n) P[n] = inb[n] ? P[n] * inv : 0.0f;  // V9 OOB = 0

  float* ap = att + (size_t)bh * 128 * HW + pos;
  for (int c = 0; c < 128; ++c) {
    const float* Vc = Vp + (size_t)c * HW;
    float acc = 0.0f;
#pragma unroll
    for (int n = 0; n < 9; ++n) acc += P[n] * Vc[npos[n]];
    ap[(size_t)c * HW] = acc;
  }
}

// ---------------------------------------------------------------------------
// out = instance_norm(A + R), one block per (batch, channel) plane of 4096.
// ---------------------------------------------------------------------------
__global__ __launch_bounds__(256) void addnorm_kernel(
    const float* __restrict__ A, const float* __restrict__ R,
    float* __restrict__ Out)
{
  const size_t base = (size_t)blockIdx.x * HW;
  const int t = threadIdx.x;

  float4 vals[4];
  float s = 0.f, sq = 0.f;
#pragma unroll
  for (int i = 0; i < 4; ++i) {
    size_t o = base + (size_t)t * 4 + (size_t)i * 1024;
    float4 a = *(const float4*)&A[o];
    float4 r = *(const float4*)&R[o];
    float4 v;
    v.x = a.x + r.x; v.y = a.y + r.y; v.z = a.z + r.z; v.w = a.w + r.w;
    vals[i] = v;
    s  += v.x + v.y + v.z + v.w;
    sq += v.x * v.x + v.y * v.y + v.z * v.z + v.w * v.w;
  }

  __shared__ float sb[256], qb[256];
  sb[t] = s; qb[t] = sq;
  __syncthreads();
  for (int o = 128; o > 0; o >>= 1) {
    if (t < o) { sb[t] += sb[t + o]; qb[t] += qb[t + o]; }
    __syncthreads();
  }
  const float mean = sb[0] * (1.0f / HW);
  const float var  = qb[0] * (1.0f / HW) - mean * mean;
  const float rstd = rsqrtf(var + 1e-5f);

#pragma unroll
  for (int i = 0; i < 4; ++i) {
    size_t o = base + (size_t)t * 4 + (size_t)i * 1024;
    float4 v = vals[i];
    float4 w;
    w.x = (v.x - mean) * rstd; w.y = (v.y - mean) * rstd;
    w.z = (v.z - mean) * rstd; w.w = (v.w - mean) * rstd;
    *(float4*)&Out[o] = w;
  }
}

// ---------------------------------------------------------------------------
extern "C" void kernel_launch(void* const* d_in, const int* in_sizes, int n_in,
                              void* d_out, int out_size, void* d_ws, size_t ws_size,
                              hipStream_t stream)
{
  (void)in_sizes; (void)n_in; (void)out_size; (void)ws_size;

  const float* x    = (const float*)d_in[0];   // [8][512][64][64]
  const float* mask = (const float*)d_in[1];   // [8][1][64][64]
  const float* qk_w = (const float*)d_in[2];   // [256][256]
  const float* qk_b = (const float*)d_in[3];
  const float* v_w  = (const float*)d_in[4];   // [128][256]
  const float* v_b  = (const float*)d_in[5];
  const float* fc_w = (const float*)d_in[6];   // [512][256]
  const float* fc_b = (const float*)d_in[7];
  const float* w1_w = (const float*)d_in[8];   // [2048][512]
  const float* w1_b = (const float*)d_in[9];
  const float* w2_w = (const float*)d_in[10];  // [512][2048]
  const float* w2_b = (const float*)d_in[11];
  float* out = (float*)d_out;

  // workspace carve-up (~384 MB)
  char* ws = (char*)d_ws;
  size_t off = 0;
  float* QKb  = (float*)(ws + off); off += (size_t)16 * 256 * HW * 4;  // 64 MB
  float* Vb   = (float*)(ws + off); off += (size_t)16 * 128 * HW * 4;  // 32 MB
  float* attb = (float*)(ws + off); off += (size_t)8  * 256 * HW * 4;  // 32 MB
  float* fcb  = (float*)(ws + off); off += (size_t)8  * 512 * HW * 4;  // 64 MB (reused for ff)
  float* attn = (float*)(ws + off); off += (size_t)8  * 512 * HW * 4;  // 64 MB
  __bf16* h1T = (__bf16*)(ws + off); off += (size_t)8 * HW * 2048 * 2; // 128 MB  [b][n][k]

  dim3 blk(256);

  // 1) QK projection: x viewed as [16][256][HW]
  gemm_wmma<256, 256, HW, float, false, false, float, false>
      <<<dim3(64, 2, 16), blk, 0, stream>>>(qk_w, x, qk_b, QKb);
  // 2) V projection
  gemm_wmma<128, 256, HW, float, false, false, float, false>
      <<<dim3(64, 1, 16), blk, 0, stream>>>(v_w, x, v_b, Vb);
  // 3) 9-neighbor attention -> att [16][128][HW] (== [8][256][HW])
  attention_kernel<<<dim3(16, 16), blk, 0, stream>>>(QKb, Vb, mask, attb);
  // 4) fc projection
  gemm_wmma<512, 256, HW, float, false, false, float, false>
      <<<dim3(64, 4, 8), blk, 0, stream>>>(fc_w, attb, fc_b, fcb);
  // 5) atten = instance_norm(fc + x)
  addnorm_kernel<<<4096, blk, 0, stream>>>(fcb, x, attn);
  // 6) h1T = leaky(w1 @ atten + b1), stored bf16 TRANSPOSED [b][n][2048]
  gemm_wmma<2048, 512, HW, float, false, true, __bf16, true>
      <<<dim3(64, 16, 8), blk, 0, stream>>>(w1_w, attn, w1_b, h1T);
  // 7) ff = w2 @ h1T + b2 (bf16 N-major input), reuse fcb
  gemm_wmma<512, 2048, HW, __bf16, true, false, float, false>
      <<<dim3(64, 4, 8), blk, 0, stream>>>(w2_w, h1T, w2_b, fcb);
  // 8) out = instance_norm(ff + atten)
  addnorm_kernel<<<4096, blk, 0, stream>>>(fcb, attn, out);
}
